// EuclideanSimilarity_60120952209550
// MI455X (gfx1250) — matrix operand order repaired and checked by reading, not compiled, not run on hardware
//
#include <hip/hip_runtime.h>

// MI455X / gfx1250, wave32. Matrix math: v_wmma_f32_16x16x32_bf16.
// B-operand staging: global_load_async_to_lds_b128 (ASYNCcnt) + s_wait_asynccnt.

typedef __attribute__((ext_vector_type(16))) __bf16 v16bf;
typedef __attribute__((ext_vector_type(8)))  __bf16 v8bf;
typedef __attribute__((ext_vector_type(8)))  float  v8f;

#define B_  8
#define L_  4096
#define D_  128
#define LH_ (L_/2)

__device__ __forceinline__ v16bf concat8(v8bf lo, v8bf hi) {
    return __builtin_shufflevector(lo, hi, 0,1,2,3,4,5,6,7,8,9,10,11,12,13,14,15);
}

// Async copy of 16 bytes global -> LDS (per-lane addresses, ASYNCcnt-tracked).
__device__ __forceinline__ void async_copy_b128(unsigned lds_addr, const void* gptr) {
    unsigned long long ga = (unsigned long long)(size_t)gptr;
    asm volatile("global_load_async_to_lds_b128 %0, %1, off"
                 :: "v"(lds_addr), "v"(ga) : "memory");
}
__device__ __forceinline__ void wait_async0() {
    asm volatile("s_wait_asynccnt 0" ::: "memory");
}

// ---------------------------------------------------------------------------
// Projection kernel: out[m][n] = sum_d A[m][d] * W[n][d] + b[n]
// mode 0: A rows = x rows            (B*L rows)   -> projb, projTb, ksq
// mode 1: A rows = pair-pooled x     (B*L/2 rows) -> qb bf16, qsq, q f32 out
// One 16-row tile per wave, 4 waves per block. W staged once in LDS as bf16.
// ---------------------------------------------------------------------------
__global__ void __launch_bounds__(128)
eucsim_proj_kernel(const float* __restrict__ x, const float* __restrict__ Wg,
                   const float* __restrict__ bg,
                   __bf16* __restrict__ outb, __bf16* __restrict__ outTb,
                   float* __restrict__ normsq, float* __restrict__ qout,
                   int mode, int ntiles)
{
    __shared__ __align__(32) __bf16 Wlds[D_ * D_];        // 32 KB, row-major [e][d]
    __shared__ __align__(32) float  tile[4][16 * D_];     // 32 KB, per-wave result tile

    const int tid  = threadIdx.x;
    const int wave = tid >> 5;
    const int lane = tid & 31;
    const int lmod = lane & 15;
    const int lhi  = lane >> 4;   // 0 or 1

    for (int i = tid; i < D_ * D_; i += 128) Wlds[i] = (__bf16)Wg[i];
    __syncthreads();

    const int gt = blockIdx.x * 4 + wave;
    if (gt >= ntiles) return;

    // ---- A fragments: 16 rows x K=128 as 4 bf16 16x16x32 A-operands ----
    // A layout: lane l holds row m=l%16; elem i -> K=(l/16)*16+i.
    v16bf aq[4];
    {
        const float* base0;
        const float* base1 = nullptr;
        if (mode == 0) {
            base0 = x + (size_t)(gt * 16 + lmod) * D_;
        } else {
            int fr   = gt * 16 + lmod;       // pooled row index
            int bidx = fr / LH_;
            int lrow = fr % LH_;
            base0 = x + ((size_t)bidx * L_ + 2 * lrow) * D_;
            base1 = base0 + D_;
        }
        #pragma unroll
        for (int c = 0; c < 4; ++c) {
            int off = 32 * c + lhi * 16;
            v8f g0 = *(const v8f*)(base0 + off);
            v8f g1 = *(const v8f*)(base0 + off + 8);
            if (mode == 1) {
                v8f h0 = *(const v8f*)(base1 + off);
                v8f h1 = *(const v8f*)(base1 + off + 8);
                g0 = 0.5f * (g0 + h0);
                g1 = 0.5f * (g1 + h1);
            }
            aq[c] = concat8(__builtin_convertvector(g0, v8bf),
                            __builtin_convertvector(g1, v8bf));
        }
    }

    // ---- GEMM: 8 n-tiles of 16 columns; K loop = 4 WMMAs each ----
    // B layout: lane l holds col n=l%16; elem i -> K=(l/16)*16+i  (contiguous
    // in d => straight 32B row loads from Wlds since B[k=d][n=e]=W[e][d]).
    #pragma unroll
    for (int nt = 0; nt < 8; ++nt) {
        v8f c0 = {};
        #pragma unroll
        for (int c = 0; c < 4; ++c) {
            const __bf16* wrow = &Wlds[(nt * 16 + lmod) * D_ + 32 * c + lhi * 16];
            v16bf bw = *(const v16bf*)wrow;
            c0 = __builtin_amdgcn_wmma_f32_16x16x32_bf16(false, aq[c], false, bw,
                                                         (short)0, c0, false, false);
        }
        float bias = bg[nt * 16 + lmod];
        // C layout: elem r -> m = r + 8*(l/16), n = l%16
        #pragma unroll
        for (int r = 0; r < 8; ++r)
            tile[wave][(r + 8 * lhi) * D_ + nt * 16 + lmod] = c0[r] + bias;
    }

    // ---- per-wave epilogue (wave-internal LDS exchange; DS in-order) ----
    if (lane < 16) {
        float s = 0.f;
        for (int j = 0; j < D_; ++j) {
            float v = tile[wave][lane * D_ + j];
            s += v * v;
        }
        normsq[gt * 16 + lane] = s;
    }
    for (int idx = lane; idx < 16 * D_; idx += 32) {
        float v = tile[wave][idx];
        outb[(size_t)gt * 16 * D_ + idx] = (__bf16)v;
        if (mode == 1) qout[(size_t)gt * 16 * D_ + idx] = v;
    }
    if (mode == 0) {
        // transposed store projT[b][d][key] for the second-GEMM B operand
        int frow = gt * 16;
        int bidx = frow / L_;
        int keyb = frow % L_;
        for (int d = lane; d < D_; d += 32) {
            v16bf t;
            #pragma unroll
            for (int m = 0; m < 16; ++m) t[m] = (__bf16)tile[wave][m * D_ + d];
            *(v16bf*)(outTb + ((size_t)bidx * D_ + d) * L_ + keyb) = t;
        }
    }
}

// ---------------------------------------------------------------------------
// Fused similarity + output GEMM, LDS-blocked for key-stream reuse.
// Block = 128 threads (4 waves) = 2 q-tiles x 2 key-halves.
//   wave w: q-tile = blockIdx*2 + (w&1), key half = w>>1.
// Each key-half group (2 waves) async-stages its 32-key proj/projT tiles into
// LDS (16 KB) once; both q-tile waves consume them (2x L2 traffic reduction).
// Per chunk: 8 qk WMMAs -> exp(-sqrt(max(qs+ks-2qk,0))) -> LDS C->A re-layout
// -> 8 accumulation WMMAs. Pairwise cross-wave f32 reduction at the end.
// ---------------------------------------------------------------------------
__global__ void __launch_bounds__(128)
eucsim_sim_kernel(const __bf16* __restrict__ qb, const __bf16* __restrict__ projb,
                  const __bf16* __restrict__ projTb,
                  const float* __restrict__ qsq, const float* __restrict__ ksq,
                  float* __restrict__ kout, float* __restrict__ vout)
{
    __shared__ __align__(32) __bf16 Pk[2][32 * D_];    // [group][key][d]   16 KB
    __shared__ __align__(32) __bf16 Pt[2][D_ * 32];    // [group][d][key]   16 KB
    __shared__ __align__(32) __bf16 Slds[4][16 * 32];  // per-wave sim tile   4 KB
    __shared__ __align__(32) float  red[2][16 * D_];   // reduction partials 16 KB

    const int tid  = threadIdx.x;
    const int wave = tid >> 5;
    const int lane = tid & 31;
    const int lmod = lane & 15;
    const int lhi  = lane >> 4;

    const int qtl     = wave & 1;                // local q-tile (0/1)
    const int keyhalf = wave >> 1;               // 0: keys 0..2047, 1: 2048..4095
    const int gtid    = qtl * 32 + lane;         // [0,64) within key-half group
    const int qt      = blockIdx.x * 2 + qtl;    // [0, 1024)
    const int b       = qt >> 7;                 // 128 q-tiles per batch

    const unsigned pk_base = (unsigned)(size_t)&Pk[keyhalf][0];
    const unsigned pt_base = (unsigned)(size_t)&Pt[keyhalf][0];

    // q A-fragments (K = 128 -> 4 fragments)
    v16bf aq[4];
    {
        const __bf16* qrow = qb + (size_t)(qt * 16 + lmod) * D_;
        #pragma unroll
        for (int c = 0; c < 4; ++c) {
            v8bf g0 = *(const v8bf*)(qrow + 32 * c + lhi * 16);
            v8bf g1 = *(const v8bf*)(qrow + 32 * c + lhi * 16 + 8);
            aq[c] = concat8(g0, g1);
        }
    }
    // ||q||^2 for the 8 rows this lane's C elements cover (m = r + 8*lhi)
    v8f qsv = *(const v8f*)(qsq + qt * 16 + 8 * lhi);

    v8f acc[8];
    {
        v8f z = {};
        #pragma unroll
        for (int nt = 0; nt < 8; ++nt) acc[nt] = z;
    }

    const float NEG_LOG2E = -1.4426950408889634f;

    for (int i = 0; i < 64; ++i) {
        const int keybase = keyhalf * 2048 + i * 32;

        // ---- async stage proj tile [32 keys][128 d] (8 KB, contiguous) ----
        {
            const __bf16* gsrc = projb + ((size_t)b * L_ + keybase) * D_;
            #pragma unroll
            for (int j = 0; j < 8; ++j) {
                int idx = gtid + j * 64;                       // [0,512) b128 chunks
                async_copy_b128(pk_base + idx * 16, (const char*)gsrc + idx * 16);
            }
        }
        // ---- async stage projT tile [128 d][32 keys] (8 KB, d-row strided) ----
        {
            #pragma unroll
            for (int j = 0; j < 8; ++j) {
                int idx  = gtid + j * 64;                      // [0,512)
                int d    = idx >> 2;
                int part = idx & 3;
                const __bf16* gsrc =
                    projTb + ((size_t)b * D_ + d) * L_ + keybase + part * 8;
                async_copy_b128(pt_base + idx * 16, gsrc);
            }
        }
        wait_async0();
        __syncthreads();   // staged tiles visible to both waves of the group

        // ---- build 16x32 similarity tile, staged through per-wave LDS ----
        #pragma unroll
        for (int h = 0; h < 2; ++h) {
            const int key16 = h * 16;
            // qk B operand: B[k=d][n=key] = proj[key][d] -> contiguous in d
            const __bf16* prow = &Pk[keyhalf][(key16 + lmod) * D_];
            v8f c0 = {};
            #pragma unroll
            for (int c = 0; c < 4; ++c) {
                v16bf bk = *(const v16bf*)(prow + 32 * c + lhi * 16);
                c0 = __builtin_amdgcn_wmma_f32_16x16x32_bf16(false, aq[c], false, bk,
                                                             (short)0, c0, false, false);
            }
            float ks = ksq[b * L_ + keybase + key16 + lmod];
            #pragma unroll
            for (int r = 0; r < 8; ++r) {
                float p    = qsv[r] + ks - 2.0f * c0[r];
                p          = fmaxf(p, 0.0f);
                float dist = __builtin_amdgcn_sqrtf(p);
                float s    = __builtin_amdgcn_exp2f(NEG_LOG2E * dist);  // exp(-dist)
                Slds[wave][(r + 8 * lhi) * 32 + key16 + lmod] = (__bf16)s;
            }
        }
        // ---- read back in A-fragment layout (wave-internal, DS in-order) ----
        v16bf as;
        {
            const __bf16* sp = &Slds[wave][lmod * 32 + lhi * 16];
            as = concat8(*(const v8bf*)sp, *(const v8bf*)(sp + 8));
        }
        // ---- acc += S @ proj : B[k=key][n=d] = Pt[n][key], contiguous keys --
        #pragma unroll
        for (int nt = 0; nt < 8; ++nt) {
            const __bf16* trow = &Pt[keyhalf][(nt * 16 + lmod) * 32 + lhi * 16];
            v16bf bp = *(const v16bf*)trow;
            acc[nt] = __builtin_amdgcn_wmma_f32_16x16x32_bf16(false, as, false, bp,
                                                              (short)0, acc[nt], false, false);
        }
        __syncthreads();   // safe to overwrite staged tiles next iteration
    }

    // ---- pairwise cross-wave reduction (key halves) ----
    if (keyhalf == 1) {
        #pragma unroll
        for (int nt = 0; nt < 8; ++nt)
            #pragma unroll
            for (int r = 0; r < 8; ++r)
                red[qtl][(r + 8 * lhi) * D_ + nt * 16 + lmod] = acc[nt][r];
    }
    __syncthreads();
    if (keyhalf == 0) {
        #pragma unroll
        for (int nt = 0; nt < 8; ++nt) {
            #pragma unroll
            for (int r = 0; r < 8; ++r) {
                int pos  = (r + 8 * lhi) * D_ + nt * 16 + lmod;
                float v  = acc[nt][r] + red[qtl][pos];
                size_t o = (size_t)qt * 16 * D_ + pos;
                kout[o] = v;
                vout[o] = v;
            }
        }
    }
}

// ---------------------------------------------------------------------------
extern "C" void kernel_launch(void* const* d_in, const int* in_sizes, int n_in,
                              void* d_out, int out_size, void* d_ws, size_t ws_size,
                              hipStream_t stream)
{
    const float* x    = (const float*)d_in[0];   // [B, L, D]
    const float* W    = (const float*)d_in[1];   // [D, D]
    const float* bias = (const float*)d_in[2];   // [D]

    const size_t qseg = (size_t)B_ * LH_ * D_;   // 2,097,152 f32 per output
    float* q_out = (float*)d_out;
    float* k_out = q_out + qseg;
    float* v_out = k_out + qseg;

    // workspace layout (all regions 256B-aligned)
    char* ws = (char*)d_ws;
    const size_t sz_projb = (size_t)B_ * L_ * D_ * 2;   // 8 MiB bf16 row-major
    const size_t sz_projT = sz_projb;                   // 8 MiB bf16 [b][d][key]
    const size_t sz_qb    = (size_t)B_ * LH_ * D_ * 2;  // 4 MiB bf16
    const size_t sz_ksq   = (size_t)B_ * L_ * 4;
    __bf16* projb  = (__bf16*)(ws);
    __bf16* projTb = (__bf16*)(ws + sz_projb);
    __bf16* qb     = (__bf16*)(ws + sz_projb + sz_projT);
    float*  ksq    = (float*)(ws + sz_projb + sz_projT + sz_qb);
    float*  qsq    = (float*)(ws + sz_projb + sz_projT + sz_qb + sz_ksq);

    // 1) projected = x @ W^T + b  (2048 tiles, 4 per block)
    eucsim_proj_kernel<<<512, 128, 0, stream>>>(x, W, bias, projb, projTb,
                                                ksq, nullptr, 0, 2048);
    // 2) q = pool2(x) @ W^T + b   (1024 tiles)
    eucsim_proj_kernel<<<256, 128, 0, stream>>>(x, W, bias, qb, nullptr,
                                                qsq, q_out, 1, 1024);
    // 3) fused distance/similarity + k = sim @ projected
    eucsim_sim_kernel<<<512, 128, 0, stream>>>(qb, projb, projTb, qsq, ksq,
                                               k_out, v_out);
}
